// forward_rnn_48558900249359
// MI455X (gfx1250) — matrix operand order repaired and checked by reading, not compiled, not run on hardware
//
#include <hip/hip_runtime.h>
#include <hip/hip_bf16.h>

typedef __bf16 bf16_t;
typedef __attribute__((ext_vector_type(16))) __bf16 v16bf;
typedef __attribute__((ext_vector_type(8)))  float  v8f;

#define HW   16384   // 128*128
#define HDIM 128

__device__ __forceinline__ float lrelu_f(float v) { return v >= 0.f ? v : 0.01f * v; }

// ---------------------------------------------------------------------------
// Logical GEMM K axis = tap*Cp + ci, tap = ky*KS+kx, Cp = pow2 >= Cin.
// Decode is pure shift/and; (ky,kx) from tap uses compile-time KS.
//
// Weight packing: fp32 OIHW -> bf16 A-fragment layout [Mtile][Kchunk][lane32][16]
// A 16x32 bf16 (contiguous k order used here; B matches): lane L (row M=L%16), half=L/16,
//   element e -> K = half*16 + e within the 32-wide chunk.
// K slots with tap>=KS*KS or ci>=Cin are packed as ZERO, which lets the conv
// kernel skip those guards on the B side (0 * garbage == 0).
// ---------------------------------------------------------------------------
template<int KS>
__global__ void pack_w_kernel(const float* __restrict__ W, bf16_t* __restrict__ Wp,
                              int Cout, int Cin, int Cp, int cShift, int Kch, int total)
{
    constexpr int KSQ = KS * KS;
    int idx = blockIdx.x * blockDim.x + threadIdx.x;
    if (idx >= total) return;
    int e    = idx & 15;
    int lane = (idx >> 4) & 31;
    int kc   = (idx >> 9) % Kch;
    int mt   = idx / (512 * Kch);
    int half = lane >> 4;
    int m    = mt * 16 + (lane & 15);
    int kk   = kc * 32 + half * 16 + e;
    int tap  = kk >> cShift;
    int ci   = kk & (Cp - 1);
    float v  = 0.f;
    if (m < Cout && tap < KSQ && ci < Cin) {
        int ky = tap / KS;
        int kx = tap - ky * KS;
        v = W[((size_t)(m * Cin + ci) * KS + ky) * KS + kx];
    }
    Wp[idx] = (bf16_t)v;
}

// ---------------------------------------------------------------------------
// Implicit-GEMM convolution on v_wmma_f32_16x16x32_bf16.
// Block: 128 threads = 4 waves. Each wave: 16(M) x 64(N) outputs as 4 subtiles
// sharing one A fragment per 32-deep K chunk (4 wmma / chunk / wave).
// B fragment (32x16 bf16): lane half h holds K = h*16 + e, col = lane%16.
// C/D v8f: element j -> M = m0 + j + 8*(lane/16), N = col.
//
// B gather is fully BRANCHLESS: addresses are clamped with cndmask and all 16
// loads issue unconditionally (clause -> single wait), invalid lanes select 0.
// K-decode is hoisted out of the 4-subtile loop (amortized 4x).
// KS==1 path: no guards at all (input temps are padded to Kch*32 channels,
// pad-channel weights are zero).
// ---------------------------------------------------------------------------
template<int KS, int STRIDE>
__global__ __launch_bounds__(128)
void wmma_conv_kernel(const float* __restrict__ in,
                      const bf16_t* __restrict__ Wp,
                      const float* __restrict__ bias,
                      const float* __restrict__ resid,
                      float* __restrict__ out,
                      int Cin, int Cp, int cShift, int Cout, int Kch,
                      int wiShift, int hwiShift,   // log2(Wi), log2(Hi*Wi)
                      int woShift, int nhwShift,   // log2(Wo), log2(Ho*Wo)
                      int pad, int lrelu)
{
    constexpr int KSQ = KS * KS;
    const int wave = threadIdx.x >> 5;
    const int lane = threadIdx.x & 31;
    const int half = lane >> 4;
    const int col  = lane & 15;
    const int m0   = blockIdx.y * 16;
    const int n0   = blockIdx.x * 256 + wave * 64;
    const int Wo   = 1 << woShift;
    const int Wi   = 1 << wiShift;
    const int Hi   = 1 << (hwiShift - wiShift);

    int nn[4], iy0[4], ix0[4], base[4];
#pragma unroll
    for (int t = 0; t < 4; ++t) {
        int n  = n0 + t * 16 + col;
        nn[t]  = n;
        int oy = n >> woShift;
        int ox = n & (Wo - 1);
        iy0[t] = oy * STRIDE - pad;
        ix0[t] = ox * STRIDE - pad;
        base[t] = (iy0[t] << wiShift) + ix0[t];
    }

    v8f bias8;
#pragma unroll
    for (int j = 0; j < 8; ++j) {
        int m = m0 + j + 8 * half;
        bias8[j] = (m < Cout) ? bias[m] : 0.f;
    }
    v8f acc[4];
#pragma unroll
    for (int t = 0; t < 4; ++t) acc[t] = bias8;

    const bf16_t* wptr = Wp + (size_t)blockIdx.y * Kch * 512 + lane * 16;
    const int kkBase = half * 16;

    for (int kc = 0; kc < Kch; ++kc) {
        v16bf a = *(const v16bf*)wptr;     // 32B-aligned A fragment, shared by 4 subtiles
        wptr += 512;
        const int kkc = kc * 32 + kkBase;

        // per-chunk K decode (independent of subtile)
        int eoff[16], kyv[16], kxv[16];
        if (KS > 1) {
#pragma unroll
            for (int e = 0; e < 16; ++e) {
                int kk  = kkc + e;
                int tap = kk >> cShift;
                int ci  = kk & (Cp - 1);
                int ky  = tap / KS;            // compile-time KS -> magic mul
                int kx  = tap - ky * KS;
                kyv[e]  = ky;
                kxv[e]  = kx;
                eoff[e] = (ci << hwiShift) + (ky << wiShift) + kx;
            }
        }

#pragma unroll
        for (int t = 0; t < 4; ++t) {
            float raw[16];
            int   okm[16];
#pragma unroll
            for (int e = 0; e < 16; ++e) {
                int addr;
                if (KS == 1) {
                    addr   = ((kkc + e) << hwiShift) + nn[t];
                    okm[e] = 1;
                } else {
                    int iy = iy0[t] + kyv[e];
                    int ix = ix0[t] + kxv[e];
                    int ok = ((unsigned)iy < (unsigned)Hi) & ((unsigned)ix < (unsigned)Wi);
                    okm[e] = ok;
                    addr   = ok ? (base[t] + eoff[e]) : 0;   // clamped, always in-bounds
                }
                raw[e] = in[addr];                           // unconditional load
            }
            v16bf b;
#pragma unroll
            for (int e = 0; e < 16; ++e) {
                float v = (KS == 1) ? raw[e] : (okm[e] ? raw[e] : 0.f);
                b[e] = (bf16_t)v;
            }
            acc[t] = __builtin_amdgcn_wmma_f32_16x16x32_bf16(false, a, false, b,
                                                             (short)0, acc[t], false, false);
        }
    }

#pragma unroll
    for (int t = 0; t < 4; ++t) {
#pragma unroll
        for (int j = 0; j < 8; ++j) {
            int m = m0 + j + 8 * half;
            if (m < Cout) {
                float v = acc[t][j];
                size_t o = ((size_t)m << nhwShift) + nn[t];
                if (resid) v += resid[o];
                if (lrelu) v = lrelu_f(v);
                out[o] = v;
            }
        }
    }
}

// ---------------------------------------------------------------------------
// ConvTranspose2d(k=3,s=2,p=1,op=1), hard-coded 20ch @64^2 -> 10ch @128^2.
// == conv over 2x lhs-dilated input with pad (1,2). Tiny; direct VALU kernel.
// ---------------------------------------------------------------------------
__global__ void convT_kernel(const float* __restrict__ in,   // [20,64,64]
                             const float* __restrict__ Wt,   // [10,20,3,3]
                             const float* __restrict__ bias,
                             float* __restrict__ out)        // [10,128,128]
{
    int idx = blockIdx.x * blockDim.x + threadIdx.x;
    if (idx >= 10 * HW) return;
    int ox = idx & 127;
    int oy = (idx >> 7) & 127;
    int co = idx >> 14;
    float acc = bias[co];
#pragma unroll
    for (int ky = 0; ky < 3; ++ky) {
        int d = oy - 1 + ky;
        if (d < 0 || d > 126 || (d & 1)) continue;
        int iy = d >> 1;
#pragma unroll
        for (int kx = 0; kx < 3; ++kx) {
            int e = ox - 1 + kx;
            if (e < 0 || e > 126 || (e & 1)) continue;
            int ix = e >> 1;
            for (int ci = 0; ci < 20; ++ci)
                acc += Wt[((co * 20 + ci) * 3 + ky) * 3 + kx]
                     * in[(ci << 12) + (iy << 6) + ix];
        }
    }
    out[idx] = acc;
}

// ---------------------------------------------------------------------------
// Elementwise helpers
// ---------------------------------------------------------------------------
__global__ void copy_kernel(const float* __restrict__ s, float* __restrict__ d, int n)
{ int i = blockIdx.x * blockDim.x + threadIdx.x; if (i < n) d[i] = s[i]; }

__global__ void zero_kernel(float* __restrict__ d, int n)
{ int i = blockIdx.x * blockDim.x + threadIdx.x; if (i < n) d[i] = 0.f; }

// suf2[i] = sum_{ii >= i+2} mask[ii], i in 0..26 (row 26 is zero)
__global__ void suffix_kernel(const float* __restrict__ mask, float* __restrict__ suf)
{
    int p = blockIdx.x * blockDim.x + threadIdx.x;
    if (p >= HW) return;
    suf[26 * HW + p] = 0.f;
    float acc = 0.f;
    for (int c = 27; c >= 2; --c) {
        acc += mask[(size_t)c * HW + p];
        suf[(size_t)(c - 2) * HW + p] = acc;
    }
}

// d1 += mask_i * xt0 ; inp = meas - d1 - suf_i * nor
__global__ void step_prep_kernel(const float* __restrict__ mask_i,
                                 const float* __restrict__ suf_i,
                                 const float* __restrict__ xt0,
                                 const float* __restrict__ meas,
                                 const float* __restrict__ nor,
                                 float* __restrict__ d1,
                                 float* __restrict__ inp)
{
    int p = blockIdx.x * blockDim.x + threadIdx.x;
    if (p >= HW) return;
    float d = d1[p] + mask_i[p] * xt0[p];
    d1[p] = d;
    inp[p] = meas[p] - d - suf_i[p] * nor[p];
}

// hc = concat(ht[20], x11[10], x12[10], x2[20])
__global__ void concat_kernel(const float* __restrict__ ht, const float* __restrict__ x11,
                              const float* __restrict__ x12, const float* __restrict__ x2,
                              float* __restrict__ hc)
{
    int idx = blockIdx.x * blockDim.x + threadIdx.x;
    if (idx >= 60 * HW) return;
    int c = idx >> 14, p = idx & (HW - 1);
    float v;
    if      (c < 20) v = ht [((size_t)c << 14) + p];
    else if (c < 30) v = x11[((size_t)(c - 20) << 14) + p];
    else if (c < 40) v = x12[((size_t)(c - 30) << 14) + p];
    else             v = x2 [((size_t)(c - 40) << 14) + p];
    hc[idx] = v;
}

// ---------------------------------------------------------------------------
// Host orchestration
// ---------------------------------------------------------------------------
struct LayerDesc { int wi; int ci, co, ks; };

// d_in param arrays start at index 6: conv_x1(12), conv_x2(12), down(10),
// up(12), h_h(6), res1(18), res2(18).  WMMA layers only (convT handled apart).
static const LayerDesc LAYERS[42] = {
    // conv_x1 [0..4]
    {6,1,16,5},{8,16,32,1},{10,32,40,3},{12,40,20,1},{14,20,20,3},
    // conv_x2 [5..9]
    {18,1,10,5},{20,10,10,1},{22,10,40,3},{24,40,20,1},{26,20,20,3},
    // down [10..14]
    {30,1,20,5},{32,20,20,1},{34,20,20,3},{36,20,40,1},{38,40,20,3},
    // up [15..20]
    {40,60,40,3},{42,40,30,1},{44,30,20,3},{46,20,20,1},{48,20,20,3},{50,20,1,1},
    // h_h [21..23]
    {52,60,30,3},{54,30,20,1},{56,20,20,3},
    // res1 [24..32]
    {58,60,60,3},{60,60,60,1},{62,60,60,3},
    {64,60,60,3},{66,60,60,1},{68,60,60,3},
    {70,60,60,3},{72,60,60,1},{74,60,60,3},
    // res2 [33..41]
    {76,60,60,3},{78,60,60,1},{80,60,60,3},
    {82,60,60,3},{84,60,60,1},{86,60,60,3},
    {88,60,60,3},{90,60,60,1},{92,60,60,3},
};

extern "C" void kernel_launch(void* const* d_in, const int* in_sizes, int n_in,
                              void* d_out, int out_size, void* d_ws, size_t ws_size,
                              hipStream_t stream)
{
    (void)in_sizes; (void)n_in; (void)out_size; (void)ws_size;
    const float* xt1  = (const float*)d_in[0];
    const float* meas = (const float*)d_in[1];
    const float* nor  = (const float*)d_in[2];
    const float* mask = (const float*)d_in[3];
    const float* h0   = (const float*)d_in[4];
    float* out = (float*)d_out;                 // [28*HW] outputs + [20*HW] ht_f

    char*  wsb = (char*)d_ws;
    size_t off = 0;

    // ---- pack all WMMA-layer weights to bf16 fragments in workspace ----
    const bf16_t* pw[42];
    const float*  pb[42];
    int pCp[42], pCs[42], pKch[42];
    for (int i = 0; i < 42; ++i) {
        const LayerDesc& l = LAYERS[i];
        int Cp = 1, cs = 0;
        while (Cp < l.ci) { Cp <<= 1; ++cs; }
        int Kch = (l.ks * l.ks * Cp + 31) / 32;
        int Mt  = (l.co + 15) / 16;
        size_t total = (size_t)Mt * Kch * 512;
        bf16_t* dst = (bf16_t*)(wsb + off);
        pw[i] = dst; pb[i] = (const float*)d_in[l.wi + 1];
        pCp[i] = Cp; pCs[i] = cs; pKch[i] = Kch;
        int blocks = (int)((total + 255) / 256);
        const float* Wsrc = (const float*)d_in[l.wi];
        if (l.ks == 1)
            pack_w_kernel<1><<<blocks, 256, 0, stream>>>(Wsrc, dst, l.co, l.ci, Cp, cs, Kch, (int)total);
        else if (l.ks == 3)
            pack_w_kernel<3><<<blocks, 256, 0, stream>>>(Wsrc, dst, l.co, l.ci, Cp, cs, Kch, (int)total);
        else
            pack_w_kernel<5><<<blocks, 256, 0, stream>>>(Wsrc, dst, l.co, l.ci, Cp, cs, Kch, (int)total);
        off = (off + total * 2 + 255) & ~(size_t)255;
    }

    // ---- activation workspace ----
    // hc/tA/tB are padded to 64 channels so the guard-free B gather can read
    // up to Kch*32 channels safely (pad-channel weights are zero).
    auto carve = [&](size_t nfl) -> float* {
        float* p = (float*)(wsb + off);
        off = (off + nfl * 4 + 255) & ~(size_t)255;
        return p;
    };
    float* hc  = carve(64 * HW);
    float* tA  = carve(64 * HW);
    float* tB  = carve(64 * HW);
    float* x11 = carve(10 * HW);
    float* x12 = carve(10 * HW);
    float* x2  = carve(20 * HW);
    float* ht  = carve(20 * HW);
    float* d1  = carve(HW);
    float* inp = carve(HW);
    float* suf = carve(27 * HW);

    auto conv = [&](int lid, const float* in, float* o, int Hin,
                    int stride, int pad, int act, const float* resid) {
        const LayerDesc& l = LAYERS[lid];
        int Ho = (Hin + 2 * pad - l.ks) / stride + 1;
        int Mt = (l.co + 15) / 16;
        int wiShift  = (Hin == 128) ? 7 : 6;
        int hwiShift = 2 * wiShift;
        int woShift  = (Ho == 128) ? 7 : 6;
        int nhwShift = 2 * woShift;
        dim3 grid((Ho * Ho) / 256, Mt);
        if (l.ks == 1)
            wmma_conv_kernel<1, 1><<<grid, 128, 0, stream>>>(in, pw[lid], pb[lid], resid, o,
                l.ci, pCp[lid], pCs[lid], l.co, pKch[lid],
                wiShift, hwiShift, woShift, nhwShift, pad, act);
        else if (l.ks == 3 && stride == 1)
            wmma_conv_kernel<3, 1><<<grid, 128, 0, stream>>>(in, pw[lid], pb[lid], resid, o,
                l.ci, pCp[lid], pCs[lid], l.co, pKch[lid],
                wiShift, hwiShift, woShift, nhwShift, pad, act);
        else if (l.ks == 3)
            wmma_conv_kernel<3, 2><<<grid, 128, 0, stream>>>(in, pw[lid], pb[lid], resid, o,
                l.ci, pCp[lid], pCs[lid], l.co, pKch[lid],
                wiShift, hwiShift, woShift, nhwShift, pad, act);
        else
            wmma_conv_kernel<5, 1><<<grid, 128, 0, stream>>>(in, pw[lid], pb[lid], resid, o,
                l.ci, pCp[lid], pCs[lid], l.co, pKch[lid],
                wiShift, hwiShift, woShift, nhwShift, pad, act);
    };

    // ---- setup ----
    suffix_kernel<<<HW / 256, 256, 0, stream>>>(mask, suf);
    copy_kernel<<<HW / 256, 256, 0, stream>>>(xt1, out, HW);          // out[:,0] = xt1
    copy_kernel<<<20 * HW / 256, 256, 0, stream>>>(h0, ht, 20 * HW);  // ht = h
    zero_kernel<<<HW / 256, 256, 0, stream>>>(d1, HW);

    // ---- x11 = conv_x1(nor_meas) ----
    conv(0, nor, tA, 128, 1, 2, 0, nullptr);   // 1->16 k5 p2
    conv(1, tA,  tB, 128, 1, 0, 1, nullptr);   // 16->32 k1 +lrelu
    conv(2, tB,  tA, 128, 2, 1, 0, nullptr);   // 32->40 k3 s2 p1 -> 64^2
    conv(3, tA,  tB,  64, 1, 0, 1, nullptr);   // 40->20 k1 +lrelu
    conv(4, tB,  tA,  64, 1, 1, 1, nullptr);   // 20->20 k3 p1 +lrelu
    convT_kernel<<<(10 * HW + 255) / 256, 256, 0, stream>>>(
        tA, (const float*)d_in[16], (const float*)d_in[17], x11);

    // ---- 27-step recurrence ----
    for (int i = 0; i < 27; ++i) {
        const float* xt = out + (size_t)i * HW;

        step_prep_kernel<<<HW / 256, 256, 0, stream>>>(
            mask + (size_t)i * HW, suf + (size_t)i * HW, xt, meas, nor, d1, inp);

        // x12 = conv_x2(inp)
        conv(5, inp, tA, 128, 1, 2, 0, nullptr);
        conv(6, tA,  tB, 128, 1, 0, 1, nullptr);
        conv(7, tB,  tA, 128, 2, 1, 0, nullptr);
        conv(8, tA,  tB,  64, 1, 0, 1, nullptr);
        conv(9, tB,  tA,  64, 1, 1, 1, nullptr);
        convT_kernel<<<(10 * HW + 255) / 256, 256, 0, stream>>>(
            tA, (const float*)d_in[28], (const float*)d_in[29], x12);

        // x2 = down(xt)
        conv(10, xt, tA, 128, 1, 2, 0, nullptr);
        conv(11, tA, tB, 128, 1, 0, 1, nullptr);
        conv(12, tB, tA, 128, 1, 1, 0, nullptr);
        conv(13, tA, tB, 128, 1, 0, 1, nullptr);
        conv(14, tB, x2, 128, 1, 1, 0, nullptr);

        // hc = concat(ht, x11, x12, x2)
        concat_kernel<<<60 * HW / 256, 256, 0, stream>>>(ht, x11, x12, x2, hc);

        // res1 + res2 : 3 residual blocks each (residual add fused in 3rd conv)
        for (int b = 0; b < 3; ++b) {
            int base = 24 + b * 3;
            conv(base + 0, hc, tA, 128, 1, 1, 1, nullptr);
            conv(base + 1, tA, tB, 128, 1, 0, 1, nullptr);
            conv(base + 2, tB, hc, 128, 1, 1, 0, hc);
        }
        for (int b = 0; b < 3; ++b) {
            int base = 33 + b * 3;
            conv(base + 0, hc, tA, 128, 1, 1, 1, nullptr);
            conv(base + 1, tA, tB, 128, 1, 0, 1, nullptr);
            conv(base + 2, tB, hc, 128, 1, 1, 0, hc);
        }

        // xt2 = up(hc) -> out slice i+1 (single channel)
        conv(15, hc, tA, 128, 1, 1, 0, nullptr);
        conv(16, tA, tB, 128, 1, 0, 1, nullptr);
        conv(17, tB, tA, 128, 1, 1, 0, nullptr);
        conv(18, tA, tB, 128, 1, 0, 1, nullptr);
        conv(19, tB, tA, 128, 1, 1, 0, nullptr);
        conv(20, tA, out + (size_t)(i + 1) * HW, 128, 1, 0, 0, nullptr);

        // ht = h_h(hc)
        conv(21, hc, tA, 128, 1, 1, 0, nullptr);
        conv(22, tA, tB, 128, 1, 0, 1, nullptr);
        conv(23, tB, ht, 128, 1, 1, 0, nullptr);
    }

    // ---- ht_f -> tail of d_out ----
    copy_kernel<<<20 * HW / 256, 256, 0, stream>>>(ht, out + 28 * (size_t)HW, 20 * HW);
}
